// HardContrastiveLoss_16277926052334
// MI455X (gfx1250) — matrix operand order repaired and checked by reading, not compiled, not run on hardware
//
#include <hip/hip_runtime.h>
#include <hip/hip_bf16.h>
#include <math.h>

// ---------------------------------------------------------------------------
// HardContrastiveLoss on MI455X (gfx1250, wave32, WMMA)
//
// S = Xn @ Xn^T  (N=4096, D=256)  via v_wmma_f32_16x16x32_f16
// per-row: exp(S/tau), mask diag+partner, exact 0.8-quantile order statistic
// (index 0.8*(4096-1)=3276, no interpolation) via bit-pattern binary search,
// hard-negative sum, -log(pos/(pos+sum)), mean over 2P terms.
// ---------------------------------------------------------------------------

typedef __attribute__((ext_vector_type(16))) _Float16 v16h;
typedef __attribute__((ext_vector_type(8)))  _Float16 v8h;
typedef __attribute__((ext_vector_type(8)))  float    v8f;

#define N_EMB   4096
#define D_EMB   256
#define P_PAIRS (N_EMB / 2)
#define INV_TAU 5.0f          // 1 / 0.2
#define EPS_N   1e-8f
#define QRANK   3276          // floor(0.8 * (N_EMB - 1)); frac is exactly 0

// ---------------------------------------------------------------------------
// Kernel 1: row L2-normalize fp32 embeddings -> f16 matrix Xh (row-major NxD)
// ---------------------------------------------------------------------------
__global__ __launch_bounds__(256)
void normalize_kernel(const float* __restrict__ E, _Float16* __restrict__ Xh) {
    const int row = blockIdx.x;
    const int t   = threadIdx.x;                 // blockDim == D_EMB == 256
    __shared__ float red[256];

    float x = E[row * D_EMB + t];
    red[t] = x * x;
    __syncthreads();
#pragma unroll
    for (int s = 128; s > 0; s >>= 1) {
        if (t < s) red[t] += red[t + s];
        __syncthreads();
    }
    float nrm = fmaxf(sqrtf(red[0]), EPS_N);
    Xh[row * D_EMB + t] = (_Float16)(x / nrm);
}

// ---------------------------------------------------------------------------
// Kernel 2: build partner[] from positive_pairs (P x 2, row-major)
// ---------------------------------------------------------------------------
__global__ __launch_bounds__(256)
void partner_kernel(const int* __restrict__ pairs, int* __restrict__ partner,
                    int nPairs) {
    int p = blockIdx.x * blockDim.x + threadIdx.x;
    if (p < nPairs) {
        int i = pairs[2 * p + 0];
        int j = pairs[2 * p + 1];
        partner[i] = j;
        partner[j] = i;
    }
}

// ---------------------------------------------------------------------------
// WMMA fragment loaders (CDNA5 ISA 16-bit layouts, wave32).
//
// A 16x32 (MxK):  lane<16: M=lane,    halfs[0..7]=K 0..7,  [8..15]=K 16..23
//                 lane>=16: M=lane-16, halfs[0..7]=K 8..15, [8..15]=K 24..31
// B 32x16 (KxN):  same K order as A: lane<16: K 0..7,16..23 ; lane>=16: K 8..15,24..31
// Since B = Xn^T, column n of B is row n of Xn -> contiguous loads.
// ---------------------------------------------------------------------------
__device__ inline v16h load_frag_a(const _Float16* __restrict__ X, int row, int k) {
    const v8h* p = (const v8h*)(X + row * D_EMB + k);
    v8h lo = p[0];   // K k..k+7
    v8h hi = p[2];   // K k+16..k+23
    v16h r;
#pragma unroll
    for (int i = 0; i < 8; ++i) { r[i] = lo[i]; r[i + 8] = hi[i]; }
    return r;
}

__device__ inline v16h load_frag_b(const _Float16* __restrict__ X, int row, int k) {
    const v8h* p = (const v8h*)(X + row * D_EMB + k);
    v8h lo = p[0];   // K k..k+7
    v8h hi = p[2];   // K k+16..k+23
    v16h r;
#pragma unroll
    for (int i = 0; i < 8; ++i) { r[i] = lo[i]; r[i + 8] = hi[i]; }
    return r;
}

// ---------------------------------------------------------------------------
// Kernel 3: S = Xh @ Xh^T (f32 out). One wave computes a 32x32 tile
// (2x2 WMMA fragments). 128x128 macro-tiles -> 16384 waves -> 2048 blocks.
// ---------------------------------------------------------------------------
__global__ __launch_bounds__(256)
void gemm_cos_kernel(const _Float16* __restrict__ Xh, float* __restrict__ S) {
    const int lane = threadIdx.x & 31;
    const int wave = (blockIdx.x << 3) | (threadIdx.x >> 5);
    const int tM   = wave >> 7;           // 0..127
    const int tN   = wave & 127;          // 0..127
    const int m0   = tM << 5;
    const int n0   = tN << 5;

    const int rSel = lane & 15;
    const int kA   = (lane >> 4) << 3;    // +0 / +8
    const int kB   = (lane >> 4) << 3;    // +0 / +8

    v8f c00 = {}, c01 = {}, c10 = {}, c11 = {};

#pragma unroll
    for (int k0 = 0; k0 < D_EMB; k0 += 32) {
        v16h a0 = load_frag_a(Xh, m0 + rSel,      k0 + kA);
        v16h a1 = load_frag_a(Xh, m0 + 16 + rSel, k0 + kA);
        v16h b0 = load_frag_b(Xh, n0 + rSel,      k0 + kB);
        v16h b1 = load_frag_b(Xh, n0 + 16 + rSel, k0 + kB);

        c00 = __builtin_amdgcn_wmma_f32_16x16x32_f16(false, a0, false, b0,
                                                     (short)0, c00, false, false);
        c01 = __builtin_amdgcn_wmma_f32_16x16x32_f16(false, a0, false, b1,
                                                     (short)0, c01, false, false);
        c10 = __builtin_amdgcn_wmma_f32_16x16x32_f16(false, a1, false, b0,
                                                     (short)0, c10, false, false);
        c11 = __builtin_amdgcn_wmma_f32_16x16x32_f16(false, a1, false, b1,
                                                     (short)0, c11, false, false);
    }

    // C/D layout: VGPR d, lane<16 -> (M=d, N=lane); lane>=16 -> (M=8+d, N=lane-16)
    const int rowOff = (lane >> 4) << 3;
    const int col    = n0 + rSel;
#pragma unroll
    for (int d = 0; d < 8; ++d) {
        const int rA = m0 + d + rowOff;
        const int rB = m0 + 16 + d + rowOff;
        S[(size_t)rA * N_EMB + col]      = c00[d];
        S[(size_t)rA * N_EMB + col + 16] = c01[d];
        S[(size_t)rB * N_EMB + col]      = c10[d];
        S[(size_t)rB * N_EMB + col + 16] = c11[d];
    }
}

// ---------------------------------------------------------------------------
// Kernel 4: per-row loss. Row resident in LDS (16KB of 320KB/WGP).
// Exact order statistic at rank QRANK via binary search on IEEE bit pattern
// (all values >= 0 -> uint order == float order), matching jnp.quantile's
// tie semantics exactly since 0.8*(N-1) has zero fractional part.
// ---------------------------------------------------------------------------
__global__ __launch_bounds__(256)
void rowloss_kernel(const float* __restrict__ S, const int* __restrict__ partner,
                    float* __restrict__ loss) {
    const int r = blockIdx.x;
    const int t = threadIdx.x;

    __shared__ float e[N_EMB];
    __shared__ float red[256];
    __shared__ float pos_sh;
    __shared__ int   cnt;

    const int pr = partner[r];

    // exp(S/tau) with mask; capture positive before masking
    for (int c = t; c < N_EMB; c += 256) {
        float ev = expf(S[(size_t)r * N_EMB + c] * INV_TAU);
        if (c == pr) pos_sh = ev;            // exactly one writer
        if (c == r || c == pr) ev = 0.0f;    // diag + positive mask
        e[c] = ev;
    }
    __syncthreads();

    // binary search: smallest u with count(e <= u) >= QRANK+1
    unsigned lo = 0u, hi = 0x7f800000u;      // [0, +inf]
    while (lo < hi) {
        unsigned mid = (lo + hi) >> 1;
        float fm = __uint_as_float(mid);
        if (t == 0) cnt = 0;
        __syncthreads();
        int local = 0;
        for (int c = t; c < N_EMB; c += 256) local += (e[c] <= fm) ? 1 : 0;
        atomicAdd(&cnt, local);
        __syncthreads();
        if (cnt >= QRANK + 1) hi = mid; else lo = mid + 1;
        __syncthreads();
    }
    const float thr = __uint_as_float(hi);

    // hard-negative sum: elements >= thr
    float local = 0.0f;
    for (int c = t; c < N_EMB; c += 256) {
        float ev = e[c];
        if (ev >= thr) local += ev;
    }
    red[t] = local;
    __syncthreads();
#pragma unroll
    for (int s = 128; s > 0; s >>= 1) {
        if (t < s) red[t] += red[t + s];
        __syncthreads();
    }
    if (t == 0) {
        float pos = pos_sh;
        loss[r] = -logf(pos / (pos + red[0]));
    }
}

// ---------------------------------------------------------------------------
// Kernel 5: sum 4096 row losses -> d_out[0] = sum / (2P)
// ---------------------------------------------------------------------------
__global__ __launch_bounds__(256)
void finalize_kernel(const float* __restrict__ loss, float* __restrict__ out) {
    const int t = threadIdx.x;
    __shared__ float red[256];
    float local = 0.0f;
    for (int i = t; i < N_EMB; i += 256) local += loss[i];
    red[t] = local;
    __syncthreads();
#pragma unroll
    for (int s = 128; s > 0; s >>= 1) {
        if (t < s) red[t] += red[t + s];
        __syncthreads();
    }
    if (t == 0) out[0] = red[0] / (float)(2 * P_PAIRS);
}

// ---------------------------------------------------------------------------
// Launcher. Workspace layout (bytes):
//   [0,            2097152) : Xh      (N*D f16, 2 MB)
//   [2097152,      2113536) : partner (N int32)
//   [2113536,      2129920) : loss    (N f32)
//   [2129920, 2129920+64MB) : S       (N*N f32)
// total ~66 MB (S also lives entirely inside the 192 MB L2).
// ---------------------------------------------------------------------------
extern "C" void kernel_launch(void* const* d_in, const int* in_sizes, int n_in,
                              void* d_out, int out_size, void* d_ws, size_t ws_size,
                              hipStream_t stream) {
    const float* emb   = (const float*)d_in[0];
    const int*   pairs = (const int*)d_in[1];
    float*       out   = (float*)d_out;

    char* ws = (char*)d_ws;
    _Float16* Xh      = (_Float16*)(ws + 0);
    int*      partner = (int*)     (ws + 2097152);
    float*    loss    = (float*)   (ws + 2113536);
    float*    S       = (float*)   (ws + 2129920);

    normalize_kernel<<<N_EMB, 256, 0, stream>>>(emb, Xh);
    partner_kernel<<<(P_PAIRS + 255) / 256, 256, 0, stream>>>(pairs, partner, P_PAIRS);
    gemm_cos_kernel<<<2048, 256, 0, stream>>>(Xh, S);
    rowloss_kernel<<<N_EMB, 256, 0, stream>>>(S, partner, loss);
    finalize_kernel<<<1, 256, 0, stream>>>(loss, out);
}
